// Encoder_1864015806581
// MI455X (gfx1250) — matrix operand order repaired and checked by reading, not compiled, not run on hardware
//
#include <hip/hip_runtime.h>
#include <hip/hip_bf16.h>
#include <math.h>

// ---------------------------------------------------------------------------
// Types for CDNA5 WMMA (wave32)
// ---------------------------------------------------------------------------
typedef __attribute__((ext_vector_type(16))) __bf16 v16bf;
typedef __attribute__((ext_vector_type(8)))  __bf16 bf16x8;
typedef __attribute__((ext_vector_type(8)))  float  v8f;

// Length chain for T=160000 with the given CONV_CFG
// 160000 -> 32001 -> 16001 -> 8001 -> 4001 -> 2001 -> 1001 -> 501
static const int  kL[8]  = {160000, 32001, 16001, 8001, 4001, 2001, 1001, 501};
#define CCH 512
#define NB  4

// ---------------------------------------------------------------------------
// CDNA5 async global->LDS copy (ASYNCcnt) via inline asm (portable across
// ROCm7.2 / amdgpu-toolchain builtin arity differences).
// lds_byte: LDS byte offset (low 32 bits of generic shared pointer).
// g       : per-lane 16B-aligned global address.
// ---------------------------------------------------------------------------
__device__ __forceinline__ void async_copy_b128(unsigned lds_byte, const void* g) {
    asm volatile("global_load_async_to_lds_b128 %0, %1, off"
                 :: "v"(lds_byte), "v"(g)
                 : "memory");
}
__device__ __forceinline__ void wait_async0() {
#if __has_builtin(__builtin_amdgcn_s_wait_asynccnt)
    __builtin_amdgcn_s_wait_asynccnt(0);
#else
    asm volatile("s_wait_asynccnt 0x0" ::: "memory");
#endif
}

// ---------------------------------------------------------------------------
// small utility kernels
// ---------------------------------------------------------------------------
__global__ void zero_stats_kernel(float* s, int n) {
    int i = threadIdx.x;
    if (i < n) s[i] = 0.0f;
}

// Reorder+convert weights: src fp32 [co][ci][j]  ->  dst bf16 [co][j][ci]
template <int KW>
__global__ void convert_w_kernel(const float* __restrict__ w,
                                 __bf16* __restrict__ dst, int n) {
    int i = blockIdx.x * blockDim.x + threadIdx.x;   // dst index
    if (i >= n) return;
    int ci = i & 511;
    int r  = i >> 9;          // co*KW + j
    int j  = r % KW;
    int co = r / KW;
    dst[i] = (__bf16)w[(co * 512 + ci) * KW + j];
}

// global mean / sumsq of the raw input (for NormalizeAudio, ddof=1)
__global__ __launch_bounds__(256) void input_stats_kernel(
    const float* __restrict__ x, int n, float* __restrict__ stats) {
    __shared__ float red[256];
    int tid = threadIdx.x;
    long i  = (long)blockIdx.x * 256 + tid;
    float v = (i < n) ? x[i] : 0.0f;
    red[tid] = v; __syncthreads();
    for (int o = 128; o > 0; o >>= 1) { if (tid < o) red[tid] += red[tid + o]; __syncthreads(); }
    if (tid == 0) atomicAdd(&stats[8], red[0]);
    __syncthreads();
    red[tid] = v * v; __syncthreads();
    for (int o = 128; o > 0; o >>= 1) { if (tid < o) red[tid] += red[tid + o]; __syncthreads(); }
    if (tid == 0) atomicAdd(&stats[9], red[0]);
}

// ---------------------------------------------------------------------------
// Layer 0: cin=1, k=10, s=5, p=5.  K tiny -> scalar fp32 kernel.
// ---------------------------------------------------------------------------
__global__ __launch_bounds__(256) void conv0_kernel(
    const float* __restrict__ x, const float* __restrict__ w,
    const float* __restrict__ bias, float* __restrict__ stats,
    __bf16* __restrict__ Y) {
    const int Lin = 160000, Lout = 32001;
    __shared__ float red[256];
    const int tid = threadIdx.x;
    const int b   = blockIdx.z;

    const float Nf   = (float)(NB * Lin);
    const float s0   = stats[8], q0 = stats[9];
    const float mean = s0 / Nf;
    const float var1 = fmaxf((q0 - s0 * s0 / Nf) / (Nf - 1.0f), 0.0f); // ddof=1
    const float inv  = 1.0f / (sqrtf(var1) + 1e-8f);

    long idx = (long)blockIdx.x * 256 + tid;            // over CCH*Lout
    float lsum = 0.0f, lsq = 0.0f;
    if (idx < (long)CCH * Lout) {
        int co = (int)(idx / Lout);
        int t  = (int)(idx % Lout);
        const float* xb = x + (long)b * Lin;
        float acc = 0.0f;
        int tin0 = t * 5 - 5;
        #pragma unroll
        for (int j = 0; j < 10; ++j) {
            int tin = tin0 + j;
            if (tin >= 0 && tin < Lin)
                acc += w[co * 10 + j] * ((xb[tin] - mean) * inv);
        }
        acc += bias[co];
        Y[(long)b * CCH * Lout + idx] = (__bf16)acc;
        lsum = acc; lsq = acc * acc;
    }
    red[tid] = lsum; __syncthreads();
    for (int o = 128; o > 0; o >>= 1) { if (tid < o) red[tid] += red[tid + o]; __syncthreads(); }
    if (tid == 0) atomicAdd(&stats[2 * b + 0], red[0]);
    __syncthreads();
    red[tid] = lsq; __syncthreads();
    for (int o = 128; o > 0; o >>= 1) { if (tid < o) red[tid] += red[tid + o]; __syncthreads(); }
    if (tid == 0) atomicAdd(&stats[2 * b + 1], red[0]);
}

// ---------------------------------------------------------------------------
// Layers 1..6: 512->512 conv as WMMA GEMM.   (stride=2, pad=1)
//   M = co (512), N = t (Lout), K = 512*KW,  kk = j*512 + ci (tap-major).
// Block: 256 threads = 8 waves (wave32). Output tile 128(co) x 64(t).
// K loop: 16 chunks of 32 ci; one staged X tile serves all KW taps.
// Double-buffered LDS; A tile copied with GLOBAL_LOAD_ASYNC_TO_LDS_B128
// (ASYNCcnt), X tile pipelined through registers -> global latency hides
// under the 4*KW WMMAs per chunk.
// Fragment layouts per CDNA5 ISA 7.12.2 (wave32):
//   A 16x32 bf16 : lane l: m=l&15; runs K=[8h..8h+7] and K=[16+8h..16+8h+7]
//   B 32x16 bf16 : lane l: n=l&15; runs K=[8h..8h+7] and K=[16+8h..16+8h+7]
//   C 16x16 f32  : vgpr r: n=l&15, m=r+8h
// ---------------------------------------------------------------------------
template <int KW>
__global__ __launch_bounds__(256) void convw_kernel(
    const __bf16* __restrict__ X, const __bf16* __restrict__ W,
    const float* __restrict__ bias, float* __restrict__ stats,
    __bf16* __restrict__ Y, int Lin, int Lout) {
    constexpr int K  = CCH * KW;
    constexpr int AG = 128 * KW * 4;          // # of 8-elem groups in A tile
    constexpr int NX = 132 * 32;              // X tile elements
    __shared__ __bf16 Alds[2][128][KW * 32];
    __shared__ __bf16 Xlds[2][132][32];
    __shared__ float  red[256];

    const int tid    = threadIdx.x;
    const int lane   = tid & 31;
    const int wv     = tid >> 5;              // 0..7
    const int half   = lane >> 4;
    const int mloc   = lane & 15;
    const int waveCo = (wv & 3) * 32;         // 0,32,64,96
    const int waveT  = (wv >> 2) * 32;        // 0,32

    const int b   = blockIdx.z;
    const int coB = blockIdx.y * 128;
    const int tB  = blockIdx.x * 64;

    const long xbase    = (long)b * CCH * Lin;
    const int  tinStart = tB * 2 - 1;         // tin for local ti=0

    // --- staging helpers -------------------------------------------------
    auto stageA_async = [&](int ci0, int bufI) {
        #pragma unroll
        for (int i = tid; i < AG; i += 256) {
            int row  = i / (KW * 4);
            int g    = i - row * (KW * 4);
            int j    = g >> 2;
            int col8 = (g & 3) * 8;
            unsigned lds = (unsigned)(unsigned long long)(const void*)
                           &Alds[bufI][row][j * 32 + col8];
            async_copy_b128(lds, &W[(long)(coB + row) * K + j * 512 + ci0 + col8]);
        }
    };
    __bf16 xv[17];
    auto loadX = [&](int ci0) {
        #pragma unroll
        for (int k2 = 0; k2 < 17; ++k2) {
            int i = tid + k2 * 256;
            __bf16 v = (__bf16)0.0f;
            if (i < NX) {
                int ti  = i % 132;            // consecutive tid -> coalesced tin
                int cl  = i / 132;
                int tin = tinStart + ti;
                if (tin >= 0 && tin < Lin)
                    v = X[xbase + (long)(ci0 + cl) * Lin + tin];
            }
            xv[k2] = v;
        }
    };
    auto storeX = [&](int bufI) {
        #pragma unroll
        for (int k2 = 0; k2 < 17; ++k2) {
            int i = tid + k2 * 256;
            if (i < NX) {
                int ti = i % 132;
                int cl = i / 132;
                Xlds[bufI][ti][cl] = xv[k2];
            }
        }
    };

    v8f c[2][2] = {};
    // --- prologue: stage chunk 0 ----------------------------------------
    stageA_async(0, 0);
    loadX(0);
    storeX(0);
    wait_async0();
    __syncthreads();

    int cur = 0;
    for (int cch = 0; cch < CCH / 32; ++cch) {
        const int ci0n = (cch + 1) * 32;
        const bool more = (ci0n < CCH);
        if (more) {
            // prefetch chunk+2 into L2 (global_prefetch_b8)
            if (ci0n + 32 < CCH) {
                int pc = ci0n + 32 + (tid >> 3);
                int pt = tinStart + (tid & 7) * 16;
                if (pt >= 0 && pt < Lin)
                    __builtin_prefetch(&X[xbase + (long)pc * Lin + pt], 0, 0);
            }
            stageA_async(ci0n, cur ^ 1);      // async copy overlaps WMMAs below
            loadX(ci0n);                      // global loads in flight
        }
        // --- compute from buffer `cur`: 4*KW WMMAs ---
        union V16 { bf16x8 h[2]; v16bf v; };
        #pragma unroll
        for (int j = 0; j < KW; ++j) {
            V16 a0, a1, b0, b1;
            const int r0 = waveCo + mloc;
            a0.h[0] = *(const bf16x8*)&Alds[cur][r0][j * 32 + 8 * half];
            a0.h[1] = *(const bf16x8*)&Alds[cur][r0][j * 32 + 16 + 8 * half];
            a1.h[0] = *(const bf16x8*)&Alds[cur][r0 + 16][j * 32 + 8 * half];
            a1.h[1] = *(const bf16x8*)&Alds[cur][r0 + 16][j * 32 + 16 + 8 * half];
            const int ti0 = 2 * (waveT + mloc) + j;
            b0.h[0] = *(const bf16x8*)&Xlds[cur][ti0][8 * half];
            b0.h[1] = *(const bf16x8*)&Xlds[cur][ti0][16 + 8 * half];
            b1.h[0] = *(const bf16x8*)&Xlds[cur][ti0 + 32][8 * half];
            b1.h[1] = *(const bf16x8*)&Xlds[cur][ti0 + 32][16 + 8 * half];

            c[0][0] = __builtin_amdgcn_wmma_f32_16x16x32_bf16(false, a0.v, false, b0.v, (short)0, c[0][0], false, false);
            c[0][1] = __builtin_amdgcn_wmma_f32_16x16x32_bf16(false, a0.v, false, b1.v, (short)0, c[0][1], false, false);
            c[1][0] = __builtin_amdgcn_wmma_f32_16x16x32_bf16(false, a1.v, false, b0.v, (short)0, c[1][0], false, false);
            c[1][1] = __builtin_amdgcn_wmma_f32_16x16x32_bf16(false, a1.v, false, b1.v, (short)0, c[1][1], false, false);
        }
        if (more) storeX(cur ^ 1);            // LDS writes for next chunk
        wait_async0();                        // A-tile async copies landed
        __syncthreads();                      // + dscnt wait inside
        cur ^= 1;
    }

    // --- epilogue: bias, bf16 store, GroupNorm stats ---
    const long ybase = (long)b * CCH * Lout;
    float lsum = 0.0f, lsq = 0.0f;
    #pragma unroll
    for (int s = 0; s < 2; ++s)
    #pragma unroll
    for (int u = 0; u < 2; ++u) {
        int t = tB + waveT + u * 16 + mloc;
        #pragma unroll
        for (int r = 0; r < 8; ++r) {
            int co  = coB + waveCo + s * 16 + r + 8 * half;
            float v = c[s][u][r] + bias[co];
            if (t < Lout) {
                Y[ybase + (long)co * Lout + t] = (__bf16)v;
                lsum += v; lsq += v * v;
            }
        }
    }
    red[tid] = lsum; __syncthreads();
    for (int o = 128; o > 0; o >>= 1) { if (tid < o) red[tid] += red[tid + o]; __syncthreads(); }
    if (tid == 0) atomicAdd(&stats[2 * b + 0], red[0]);
    __syncthreads();
    red[tid] = lsq; __syncthreads();
    for (int o = 128; o > 0; o >>= 1) { if (tid < o) red[tid] += red[tid + o]; __syncthreads(); }
    if (tid == 0) atomicAdd(&stats[2 * b + 1], red[0]);
}

// ---------------------------------------------------------------------------
// GroupNorm(1 group over C,L per batch) + exact GELU, in place on bf16 buffer
// ---------------------------------------------------------------------------
__global__ __launch_bounds__(256) void gn_gelu_kernel(
    __bf16* __restrict__ buf, const float* __restrict__ gnw,
    const float* __restrict__ gnb, const float* __restrict__ stats, int Lout) {
    const int b  = blockIdx.z;
    const long n = (long)CCH * Lout;
    const float Nf   = (float)n;
    const float s    = stats[2 * b + 0];
    const float q    = stats[2 * b + 1];
    const float mean = s / Nf;
    const float var  = fmaxf(q / Nf - mean * mean, 0.0f);   // ddof=0
    const float inv  = rsqrtf(var + 1e-5f);
    __bf16* base = buf + (long)b * n;
    for (long i = (long)blockIdx.x * blockDim.x + threadIdx.x; i < n;
         i += (long)gridDim.x * blockDim.x) {
        int c   = (int)(i / Lout);
        float v = (float)base[i];
        v = (v - mean) * inv * gnw[c] + gnb[c];
        v = 0.5f * v * (1.0f + erff(v * 0.70710678118654752f));  // exact GELU
        base[i] = (__bf16)v;
    }
}

// ---------------------------------------------------------------------------
// time_masking + fp32 output
// ---------------------------------------------------------------------------
__global__ __launch_bounds__(256) void mask_kernel(
    const __bf16* __restrict__ act, const float* __restrict__ memb,
    const int* __restrict__ starts, float* __restrict__ out) {
    const int F = 501;
    long total = (long)NB * CCH * F;
    long i = (long)blockIdx.x * 256 + threadIdx.x;
    if (i >= total) return;
    int t = (int)(i % F);
    int c = (int)((i / F) % CCH);
    int b = (int)(i / ((long)F * CCH));
    bool m = false;
    #pragma unroll
    for (int k = 0; k < 3; ++k) {
        int s = starts[b * 3 + k];
        m = m || (t >= s && t < s + 10);
    }
    out[i] = m ? memb[c] : (float)act[i];
}

// ---------------------------------------------------------------------------
// host launcher
// ---------------------------------------------------------------------------
extern "C" void kernel_launch(void* const* d_in, const int* in_sizes, int n_in,
                              void* d_out, int out_size, void* d_ws, size_t ws_size,
                              hipStream_t stream) {
    const float* x = (const float*)d_in[0];
    const float* wP[7], *bP[7], *gwP[7], *gbP[7];
    for (int i = 0; i < 7; ++i) {
        wP[i]  = (const float*)d_in[1 + 4 * i + 0];
        bP[i]  = (const float*)d_in[1 + 4 * i + 1];
        gwP[i] = (const float*)d_in[1 + 4 * i + 2];
        gbP[i] = (const float*)d_in[1 + 4 * i + 3];
    }
    const float* memb   = (const float*)d_in[29];
    const int*   starts = (const int*)d_in[30];
    float* out = (float*)d_out;

    // workspace layout
    float* stats = (float*)d_ws;                              // 16 floats
    size_t off   = 256;
    __bf16* wbf[7];
    size_t wElems[7] = {0, 512 * 512 * 3, 512 * 512 * 3, 512 * 512 * 3,
                        512 * 512 * 3, 512 * 512 * 2, 512 * 512 * 2};
    for (int i = 1; i < 7; ++i) { wbf[i] = (__bf16*)((char*)d_ws + off); off += wElems[i] * 2; }
    off = (off + 255) & ~(size_t)255;
    __bf16* bufA = (__bf16*)((char*)d_ws + off);              // max: L0 out
    off += (size_t)NB * CCH * kL[1] * 2;
    off = (off + 255) & ~(size_t)255;
    __bf16* bufB = (__bf16*)((char*)d_ws + off);              // max: L1 out

    // 1) convert + tap-major reorder of 512->512 weights
    for (int i = 1; i < 7; ++i) {
        int n = (int)wElems[i];
        if (i <= 4) convert_w_kernel<3><<<(n + 255) / 256, 256, 0, stream>>>(wP[i], wbf[i], n);
        else        convert_w_kernel<2><<<(n + 255) / 256, 256, 0, stream>>>(wP[i], wbf[i], n);
    }
    // 2) zero all stats, then input normalization stats
    zero_stats_kernel<<<1, 32, 0, stream>>>(stats, 16);
    {
        int n = NB * kL[0];
        input_stats_kernel<<<(n + 255) / 256, 256, 0, stream>>>(x, n, stats);
    }
    // 3) layer 0 (scalar) -> bufA, then GN+GELU
    {
        long n = (long)CCH * kL[1];
        dim3 g((unsigned)((n + 255) / 256), 1, NB);
        conv0_kernel<<<g, 256, 0, stream>>>(x, wP[0], bP[0], stats, bufA);
        gn_gelu_kernel<<<dim3(2048, 1, NB), 256, 0, stream>>>(bufA, gwP[0], gbP[0], stats, kL[1]);
    }
    // 4) layers 1..6 (WMMA), ping-pong bufA <-> bufB
    __bf16* src = bufA;
    __bf16* dst = bufB;
    for (int i = 1; i < 7; ++i) {
        int Lin = kL[i], Lout = kL[i + 1];
        zero_stats_kernel<<<1, 32, 0, stream>>>(stats, 8);
        dim3 g((unsigned)((Lout + 63) / 64), CCH / 128, NB);
        if (i <= 4)
            convw_kernel<3><<<g, 256, 0, stream>>>(src, wbf[i], bP[i], stats, dst, Lin, Lout);
        else
            convw_kernel<2><<<g, 256, 0, stream>>>(src, wbf[i], bP[i], stats, dst, Lin, Lout);
        gn_gelu_kernel<<<dim3(1024, 1, NB), 256, 0, stream>>>(dst, gwP[i], gbP[i], stats, Lout);
        __bf16* tmp = src; src = dst; dst = tmp;
    }
    // final activation is in `src` (layer-6 output), length 501
    {
        long total = (long)NB * CCH * kL[7];
        mask_kernel<<<(unsigned)((total + 255) / 256), 256, 0, stream>>>(src, memb, starts, out);
    }
}
